// MultiheadAttention_17102559773260
// MI455X (gfx1250) — matrix operand order repaired and checked by reading, not compiled, not run on hardware
//
#include <hip/hip_runtime.h>

#define D_MODEL 1024
#define SEQ     2048
#define BATCH   4
#define NHEAD   16
#define DHEAD   64

typedef __attribute__((ext_vector_type(16))) _Float16     v16h;
typedef __attribute__((ext_vector_type(8)))  float        v8f;
typedef __attribute__((ext_vector_type(4)))  unsigned int v4u;
typedef __attribute__((ext_vector_type(8)))  int          v8i;
typedef __attribute__((ext_vector_type(4)))  int          v4i;

// ---------------------------------------------------------------------------
// CDNA5 WMMA fragment layouts (CDNA5 ISA, wave32):
//  A (16x32 f16): lane L -> row = L&15, half = L>>4;
//    v16h elem (2j+p): K = (j>>2)*16 + half*8 + (j&3)*2 + p
//  B (32x16 f16): lane L -> col = L&15, half = L>>4;
//    v16h elem (2j+p): K = (j>>2)*16 + half*8 + (j&3)*2 + p (same as A)
//  C/D (16x16 f32): elem i -> row = i + 8*half, col = L&15
// ---------------------------------------------------------------------------

__device__ __forceinline__ v8f wmma_f16(v16h a, v16h b, v8f c) {
  return __builtin_amdgcn_wmma_f32_16x16x32_f16(
      /*neg_a=*/false, a, /*neg_b=*/false, b,
      /*c_mod=*/(short)0, c, /*reuse_a=*/false, /*reuse_b=*/false);
}

__device__ __forceinline__ v16h load_a_frag(const _Float16* row_ptr, int lh) {
  v16h a;
#pragma unroll
  for (int j = 0; j < 8; ++j) {
    const int k0 = ((j >> 2) * 16) + lh * 8 + ((j & 3) * 2);
    a[2 * j]     = row_ptr[k0];
    a[2 * j + 1] = row_ptr[k0 + 1];
  }
  return a;
}

__device__ __forceinline__ v16h load_b_frag(const _Float16* col_ptr, int lh) {
  v16h b;
#pragma unroll
  for (int j = 0; j < 8; ++j) {
    const int k0 = ((j >> 2) * 16) + lh * 8 + ((j & 3) * 2);
    b[2 * j]     = col_ptr[k0];
    b[2 * j + 1] = col_ptr[k0 + 1];
  }
  return b;
}

// Issue a TDM 2D-tile load (rows x 64 f16, row stride 64) into LDS.
// D# per CDNA5 ISA. Wave-level op; caller fences
// with s_wait_tensorcnt and a workgroup barrier.
__device__ __forceinline__ void tdm_load_tile_f16(
    unsigned lds_addr, const void* gptr, int rows, int tensor_rows) {
  const unsigned long long ga = (unsigned long long)gptr;
  v4u g0;
  g0[0] = 1u;                                    // count=1, is_restore=0
  g0[1] = lds_addr;                              // LDS byte address
  g0[2] = (unsigned)(ga & 0xffffffffull);        // global_addr[31:0]
  g0[3] = (unsigned)((ga >> 32) & 0x01ffffffu)   // global_addr[56:32]
          | 0x80000000u;                         // type=2 ("image")
  v8i g1;
  g1[0] = (int)(1u << 16);                       // data_size=1 (2 bytes)
  g1[1] = (int)(64u << 16);                      // tensor_dim0=64 (bits 79:48 lo)
  g1[2] = (int)(((unsigned)tensor_rows) << 16);  // tensor_dim1 lo16 (bits 111:80)
  g1[3] = (int)(64u << 16);                      // tile_dim0=64 (bits 127:112)
  g1[4] = rows;                                  // tile_dim1 (bits 143:128)
  g1[5] = 64;                                    // tensor_dim0_stride lo32
  g1[6] = 0;
  g1[7] = 0;
  v4i gz = {0, 0, 0, 0};
#if defined(__clang_major__) && (__clang_major__ >= 23)
  v8i gz8 = {0, 0, 0, 0, 0, 0, 0, 0};
  __builtin_amdgcn_tensor_load_to_lds(g0, g1, gz, gz, gz8, 0);
#else
  __builtin_amdgcn_tensor_load_to_lds(g0, g1, gz, gz, 0);
#endif
}

// ---------------------------------------------------------------------------
// Projection GEMM: Y = X(fp32)[8192,1024] @ W(fp32)[1024,1024] + b, output f16
// scattered into head-split [B, H, S, Dh]. CTA tile 128x128, 8 waves (4x2),
// per-wave 32x64 -> 8 v_wmma per K-chunk.
// ---------------------------------------------------------------------------
__global__ __launch_bounds__(256) void proj_gemm_kernel(
    const float* __restrict__ X, const float* __restrict__ W,
    const float* __restrict__ bias, _Float16* __restrict__ Y) {
  __shared__ __align__(16) _Float16 As[128 * 32];  // [m][k] k-major
  __shared__ __align__(16) _Float16 Bs[128 * 32];  // [n][k] k-major (transposed)

  const int tid  = threadIdx.x;
  const int lane = tid & 31, wave = tid >> 5;
  const int lh = lane >> 4, l15 = lane & 15;
  const int waveRow = wave >> 1, waveCol = wave & 1;
  const int m0 = blockIdx.y * 128;
  const int n0 = blockIdx.x * 128;

  v8f acc[2][4];
#pragma unroll
  for (int r = 0; r < 2; ++r)
#pragma unroll
    for (int c = 0; c < 4; ++c)
#pragma unroll
      for (int i = 0; i < 8; ++i) acc[r][c][i] = 0.0f;

  for (int k0 = 0; k0 < D_MODEL; k0 += 32) {
    // Stage X tile 128x32 fp32 -> f16
#pragma unroll
    for (int it = 0; it < 4; ++it) {
      const int s = tid + it * 256;
      const int row = s >> 3, cg = (s & 7) * 4;
      const float4 xv = *reinterpret_cast<const float4*>(
          X + (size_t)(m0 + row) * D_MODEL + k0 + cg);
      As[row * 32 + cg + 0] = (_Float16)xv.x;
      As[row * 32 + cg + 1] = (_Float16)xv.y;
      As[row * 32 + cg + 2] = (_Float16)xv.z;
      As[row * 32 + cg + 3] = (_Float16)xv.w;
    }
    // Stage W tile 32x128 fp32 -> f16, transposed to [n][k]
#pragma unroll
    for (int it = 0; it < 4; ++it) {
      const int s = tid + it * 256;
      const int kk = s >> 5, cg = (s & 31) * 4;
      const float4 wv = *reinterpret_cast<const float4*>(
          W + (size_t)(k0 + kk) * D_MODEL + n0 + cg);
      Bs[(cg + 0) * 32 + kk] = (_Float16)wv.x;
      Bs[(cg + 1) * 32 + kk] = (_Float16)wv.y;
      Bs[(cg + 2) * 32 + kk] = (_Float16)wv.z;
      Bs[(cg + 3) * 32 + kk] = (_Float16)wv.w;
    }
    __syncthreads();

    v16h af[2];
#pragma unroll
    for (int r = 0; r < 2; ++r)
      af[r] = load_a_frag(&As[(waveRow * 32 + r * 16 + l15) * 32], lh);
#pragma unroll
    for (int c = 0; c < 4; ++c) {
      const v16h bf = load_b_frag(&Bs[(waveCol * 64 + c * 16 + l15) * 32], lh);
#pragma unroll
      for (int r = 0; r < 2; ++r) acc[r][c] = wmma_f16(af[r], bf, acc[r][c]);
    }
    __syncthreads();
  }

  // Epilogue: bias, convert to f16, scatter into [B, H, S, Dh]
#pragma unroll
  for (int r = 0; r < 2; ++r)
#pragma unroll
    for (int c = 0; c < 4; ++c) {
      const int n  = n0 + waveCol * 64 + c * 16 + l15;
      const float bn = bias[n];
      const int hh = n >> 6, dd = n & 63;
#pragma unroll
      for (int i = 0; i < 8; ++i) {
        const int m  = m0 + waveRow * 32 + r * 16 + i + 8 * lh;
        const int bb = m >> 11, ss = m & 2047;
        Y[(((size_t)bb * NHEAD + hh) * SEQ + ss) * DHEAD + dd] =
            (_Float16)(acc[r][c][i] + bn);
      }
    }
}

// ---------------------------------------------------------------------------
// Flash-attention: one CTA per (b, h, 64-query block); 4 waves x 16 q-rows,
// online softmax over 32-key blocks. K-block staged by the Tensor Data Mover.
// ---------------------------------------------------------------------------
__global__ __launch_bounds__(128) void attn_kernel(
    const _Float16* __restrict__ Qh, const _Float16* __restrict__ Kh,
    const _Float16* __restrict__ Vh, _Float16* __restrict__ Oa) {
  const int blk = blockIdx.x;
  const int qb  = blk & (SEQ / 64 - 1);
  const int bh  = blk >> 5;
  const int bb  = bh / NHEAD, hh = bh % NHEAD;
  const size_t base = (size_t)bh * SEQ * DHEAD;

  const int tid  = threadIdx.x;
  const int lane = tid & 31, wave = tid >> 5;
  const int lh = lane >> 4, l15 = lane & 15;

  __shared__ __align__(16) _Float16 Ks[32 * 64];     // [key][d] (TDM target)
  __shared__ __align__(16) _Float16 Vt[64 * 32];     // [d][key]
  __shared__ __align__(16) _Float16 Ps[4][16 * 32];  // per-wave P tile

  // Q fragments (Dh=64 -> 2 chunks of K=32), pre-scaled by 1/sqrt(Dh)=0.125
  const int qrow = qb * 64 + wave * 16 + l15;
  const _Float16* qptr = Qh + base + (size_t)qrow * DHEAD;
  v16h qa[2];
#pragma unroll
  for (int ck = 0; ck < 2; ++ck)
#pragma unroll
    for (int j = 0; j < 8; ++j) {
      const int k0 = ck * 32 + ((j >> 2) * 16) + lh * 8 + ((j & 3) * 2);
      qa[ck][2 * j]     = (_Float16)((float)qptr[k0] * 0.125f);
      qa[ck][2 * j + 1] = (_Float16)((float)qptr[k0 + 1] * 0.125f);
    }

  float m_i[8], l_i[8];
  v8f Oacc[4];
#pragma unroll
  for (int i = 0; i < 8; ++i) { m_i[i] = -1e30f; l_i[i] = 0.0f; }
#pragma unroll
  for (int c = 0; c < 4; ++c)
#pragma unroll
    for (int i = 0; i < 8; ++i) Oacc[c][i] = 0.0f;

  for (int kb = 0; kb < SEQ / 32; ++kb) {
    // --- K block via Tensor Data Mover: 32x64 f16 tile -> Ks --------------
    if (wave == 0) {
      tdm_load_tile_f16((unsigned)(size_t)&Ks[0],
                        Kh + base + (size_t)kb * 32 * DHEAD,
                        /*rows=*/32, /*tensor_rows=*/SEQ);
    }
    // --- V block transposed staging (manual: TDM cannot transpose) --------
    {
      const int r  = tid >> 2;          // key row 0..31
      const int dg = (tid & 3) * 16;    // d group
      const _Float16* vg = Vh + base + (size_t)(kb * 32 + r) * DHEAD + dg;
#pragma unroll
      for (int j = 0; j < 16; ++j) Vt[(dg + j) * 32 + r] = vg[j];
    }
    {
      const int kbn = (kb + 1 < SEQ / 32) ? kb + 1 : kb;
      __builtin_prefetch(Vh + base + (size_t)kbn * 32 * DHEAD + tid * 16, 0, 3);
    }
    if (wave == 0) __builtin_amdgcn_s_wait_tensorcnt((short)0);
    __syncthreads();

    // Scores: 16 queries x 32 keys; K-dim = Dh = 2x32
    v8f sc[2];
#pragma unroll
    for (int t = 0; t < 2; ++t) {
      const int key = t * 16 + l15;
      v8f z;
#pragma unroll
      for (int i = 0; i < 8; ++i) z[i] = 0.0f;
      const v16h bk0 = load_b_frag(&Ks[key * 64 + 0], lh);
      sc[t] = wmma_f16(qa[0], bk0, z);
      const v16h bk1 = load_b_frag(&Ks[key * 64 + 32], lh);
      sc[t] = wmma_f16(qa[1], bk1, sc[t]);
    }

    // Online softmax; row = i + 8*lh, cols striped over lanes of the half
    float scale_f[8];
#pragma unroll
    for (int i = 0; i < 8; ++i) {
      float mx = fmaxf(sc[0][i], sc[1][i]);
#pragma unroll
      for (int m = 1; m < 16; m <<= 1) mx = fmaxf(mx, __shfl_xor(mx, m, 32));
      const float mnew = fmaxf(m_i[i], mx);
      const float p0 = __expf(sc[0][i] - mnew);
      const float p1 = __expf(sc[1][i] - mnew);
      float rs = p0 + p1;
#pragma unroll
      for (int m = 1; m < 16; m <<= 1) rs += __shfl_xor(rs, m, 32);
      const float corr = __expf(m_i[i] - mnew);
      l_i[i] = l_i[i] * corr + rs;
      m_i[i] = mnew;
      scale_f[i] = corr;
      sc[0][i] = p0;
      sc[1][i] = p1;
    }
#pragma unroll
    for (int c = 0; c < 4; ++c)
#pragma unroll
      for (int i = 0; i < 8; ++i) Oacc[c][i] *= scale_f[i];

    // P tile: D-layout f32 -> A-layout f16 via per-wave LDS bounce
    _Float16* P = Ps[wave];
#pragma unroll
    for (int t = 0; t < 2; ++t)
#pragma unroll
      for (int i = 0; i < 8; ++i)
        P[(i + 8 * lh) * 32 + t * 16 + l15] = (_Float16)sc[t][i];
    const v16h pa = load_a_frag(&P[l15 * 32], lh);

    // O += P @ V
#pragma unroll
    for (int c = 0; c < 4; ++c) {
      const v16h vb = load_b_frag(&Vt[(c * 16 + l15) * 32], lh);
      Oacc[c] = wmma_f16(pa, vb, Oacc[c]);
    }
    __syncthreads();
  }

  // Normalize and store to [B, S, H*Dh] f16
#pragma unroll
  for (int c = 0; c < 4; ++c) {
    const int d = c * 16 + l15;
#pragma unroll
    for (int i = 0; i < 8; ++i) {
      const int row = i + 8 * lh;
      const int s   = qb * 64 + wave * 16 + row;
      const float v = Oacc[c][i] / l_i[i];
      Oa[((size_t)bb * SEQ + s) * D_MODEL + hh * DHEAD + d] = (_Float16)v;
    }
  }
}

// ---------------------------------------------------------------------------
// Output projection: out = Xh(f16)[8192,1024] @ W(fp32)[1024,1024] + b -> fp32.
// A-tile staged with async global->LDS (ASYNCcnt), CTA tile 128x128.
// ---------------------------------------------------------------------------
__global__ __launch_bounds__(256) void out_gemm_kernel(
    const _Float16* __restrict__ Xh, const float* __restrict__ W,
    const float* __restrict__ bias, float* __restrict__ out) {
  __shared__ __align__(16) _Float16 As[128 * 32];
  __shared__ __align__(16) _Float16 Bs[128 * 32];

  const int tid  = threadIdx.x;
  const int lane = tid & 31, wave = tid >> 5;
  const int lh = lane >> 4, l15 = lane & 15;
  const int waveRow = wave >> 1, waveCol = wave & 1;
  const int m0 = blockIdx.y * 128;
  const int n0 = blockIdx.x * 128;

  const _Float16* xbase = Xh + (size_t)m0 * D_MODEL;  // uniform -> SGPR pair

  v8f acc[2][4];
#pragma unroll
  for (int r = 0; r < 2; ++r)
#pragma unroll
    for (int c = 0; c < 4; ++c)
#pragma unroll
      for (int i = 0; i < 8; ++i) acc[r][c][i] = 0.0f;

  for (int k0 = 0; k0 < D_MODEL; k0 += 32) {
    // Stage X tile 128x32 f16 via async global->LDS b128 copies
#pragma unroll
    for (int it = 0; it < 2; ++it) {
      const int s = tid + it * 256;
      const int row = s >> 2, cg = (s & 3) * 8;
      const unsigned ldsb = (unsigned)(size_t)&As[row * 32 + cg];
      const unsigned goff = (unsigned)((row * D_MODEL + k0 + cg) * 2);
      asm volatile("global_load_async_to_lds_b128 %0, %1, %2"
                   :: "v"(ldsb), "v"(goff), "s"(xbase) : "memory");
    }
    // Stage W tile 32x128 fp32 -> f16 transposed [n][k]
#pragma unroll
    for (int it = 0; it < 4; ++it) {
      const int s = tid + it * 256;
      const int kk = s >> 5, cg = (s & 31) * 4;
      const float4 wv = *reinterpret_cast<const float4*>(
          W + (size_t)(k0 + kk) * D_MODEL + n0 + cg);
      Bs[(cg + 0) * 32 + kk] = (_Float16)wv.x;
      Bs[(cg + 1) * 32 + kk] = (_Float16)wv.y;
      Bs[(cg + 2) * 32 + kk] = (_Float16)wv.z;
      Bs[(cg + 3) * 32 + kk] = (_Float16)wv.w;
    }
    asm volatile("s_wait_asynccnt 0x0" ::: "memory");
    __syncthreads();

    v16h af[2];
#pragma unroll
    for (int r = 0; r < 2; ++r)
      af[r] = load_a_frag(&As[(waveRow * 32 + r * 16 + l15) * 32], lh);
#pragma unroll
    for (int c = 0; c < 4; ++c) {
      const v16h bf = load_b_frag(&Bs[(waveCol * 64 + c * 16 + l15) * 32], lh);
#pragma unroll
      for (int r = 0; r < 2; ++r) acc[r][c] = wmma_f16(af[r], bf, acc[r][c]);
    }
    __syncthreads();
  }

#pragma unroll
  for (int r = 0; r < 2; ++r)
#pragma unroll
    for (int c = 0; c < 4; ++c) {
      const int n = n0 + waveCol * 64 + c * 16 + l15;
      const float bn = bias[n];
#pragma unroll
      for (int i = 0; i < 8; ++i) {
        const int m = m0 + waveRow * 32 + r * 16 + i + 8 * lh;
        out[(size_t)m * D_MODEL + n] = acc[r][c][i] + bn;
      }
    }
}

// ---------------------------------------------------------------------------
extern "C" void kernel_launch(void* const* d_in, const int* in_sizes, int n_in,
                              void* d_out, int out_size, void* d_ws, size_t ws_size,
                              hipStream_t stream) {
  (void)in_sizes; (void)n_in; (void)out_size; (void)ws_size;

  const float* q   = (const float*)d_in[0];
  const float* k   = (const float*)d_in[1];
  const float* v   = (const float*)d_in[2];
  const float* w_q = (const float*)d_in[3];
  const float* b_q = (const float*)d_in[4];
  const float* w_k = (const float*)d_in[5];
  const float* b_k = (const float*)d_in[6];
  const float* w_v = (const float*)d_in[7];
  const float* b_v = (const float*)d_in[8];
  const float* w_o = (const float*)d_in[9];
  const float* b_o = (const float*)d_in[10];
  float* out = (float*)d_out;

  // Workspace: 4 f16 buffers of B*H*S*Dh = 8,388,608 elements (16 MB each)
  const size_t ELEMS = (size_t)BATCH * NHEAD * SEQ * DHEAD;
  char* ws = (char*)d_ws;
  _Float16* qh = (_Float16*)(ws);
  _Float16* kh = (_Float16*)(ws + 2 * ELEMS);
  _Float16* vh = (_Float16*)(ws + 4 * ELEMS);
  _Float16* oa = (_Float16*)(ws + 6 * ELEMS);

  const dim3 gemmGrid(D_MODEL / 128, (BATCH * SEQ) / 128);  // 8 x 64
  proj_gemm_kernel<<<gemmGrid, 256, 0, stream>>>(q, w_q, b_q, qh);
  proj_gemm_kernel<<<gemmGrid, 256, 0, stream>>>(k, w_k, b_k, kh);
  proj_gemm_kernel<<<gemmGrid, 256, 0, stream>>>(v, w_v, b_v, vh);

  attn_kernel<<<BATCH * NHEAD * (SEQ / 64), 128, 0, stream>>>(qh, kh, vh, oa);

  out_gemm_kernel<<<gemmGrid, 256, 0, stream>>>(oa, w_o, b_o, out);
}
